// CompleteFeedForwardExpert_40372692582707
// MI455X (gfx1250) — compile-verified, matrix operand order checked by reading, not run on hardware
//
#include <hip/hip_runtime.h>
#include <math.h>

// Problem shape (fixed by the reference): B=4, S=4096, D=1024, H=4096, R=8
#define N_TOK   16384
#define D_IN    1024
#define H_DIM   4096
#define SCALE_F 2.0f     // lora_alpha / lora_r = 16/8
#define LN_EPS  1e-5f

typedef __bf16 bf16_t;
typedef __attribute__((ext_vector_type(4)))  __bf16 v4bf;
typedef __attribute__((ext_vector_type(8)))  __bf16 v8bf;
typedef __attribute__((ext_vector_type(16))) __bf16 v16bf;
typedef __attribute__((ext_vector_type(8)))  float  v8f;
typedef __attribute__((ext_vector_type(4)))  int    v4i;

#if __has_builtin(__builtin_amdgcn_global_load_async_to_lds_b128) && \
    __has_builtin(__builtin_amdgcn_s_wait_asynccnt)
#define USE_ASYNC_LDS 1
typedef __attribute__((address_space(1))) v4i* gv4i_p;   // global int4*
typedef __attribute__((address_space(3))) v4i* lv4i_p;   // LDS int4*
#else
#define USE_ASYNC_LDS 0
#endif

struct B32x8 { uint4 lo; uint4 hi; };   // 32 B, bit_cast carrier for v16bf

__device__ __forceinline__ float gelu_exact(float v) {
    return 0.5f * v * (1.0f + erff(v * 0.70710678118654752440f));
}

// ---------------------------------------------------------------------------
// Kernel 1: cast x -> bf16 and compute t_up[n,r] = sum_d x[n,d] * A_up[r,d]
// ---------------------------------------------------------------------------
__global__ void __launch_bounds__(256) prep_x_kernel(
    const float* __restrict__ x, const float* __restrict__ A_up,
    bf16_t* __restrict__ xb, float* __restrict__ t_up) {
    const int token = blockIdx.x;
    const int tid   = threadIdx.x;
    const int d0    = tid * 4;

    const float4 xv = *(const float4*)(x + (size_t)token * D_IN + d0);
    v4bf xbv;
    xbv[0] = (__bf16)xv.x; xbv[1] = (__bf16)xv.y;
    xbv[2] = (__bf16)xv.z; xbv[3] = (__bf16)xv.w;
    *(v4bf*)(xb + (size_t)token * D_IN + d0) = xbv;

    float part[8];
#pragma unroll
    for (int r = 0; r < 8; ++r) {
        const float4 av = *(const float4*)(A_up + (size_t)r * D_IN + d0);
        part[r] = xv.x * av.x + xv.y * av.y + xv.z * av.z + xv.w * av.w;
    }

    __shared__ float red[8][257];
#pragma unroll
    for (int r = 0; r < 8; ++r) red[r][tid] = part[r];
    __syncthreads();

    const int w = tid >> 5;        // 8 waves of 32 (wave32)
    const int lane = tid & 31;
    float s = 0.f;
#pragma unroll
    for (int i = 0; i < 8; ++i) s += red[w][lane + i * 32];
#pragma unroll
    for (int off = 16; off > 0; off >>= 1) s += __shfl_down(s, off, 32);
    if (lane == 0) t_up[(size_t)token * 8 + w] = s;
}

// ---------------------------------------------------------------------------
// Kernel 2: cast Wi -> bf16 (elementwise, 8 elems / thread)
// ---------------------------------------------------------------------------
__global__ void __launch_bounds__(256) conv_bf16_kernel(
    const float* __restrict__ in, bf16_t* __restrict__ out) {
    const size_t i = ((size_t)blockIdx.x * 256 + threadIdx.x) * 8;
    const float4 a = *(const float4*)(in + i);
    const float4 b = *(const float4*)(in + i + 4);
    v8bf o;
    o[0] = (__bf16)a.x; o[1] = (__bf16)a.y; o[2] = (__bf16)a.z; o[3] = (__bf16)a.w;
    o[4] = (__bf16)b.x; o[5] = (__bf16)b.y; o[6] = (__bf16)b.z; o[7] = (__bf16)b.w;
    *(v8bf*)(out + i) = o;
}

// ---------------------------------------------------------------------------
// Kernel 3: fused GEMM + bias + exact GELU + LoRA-up + rank-8 A_down contract.
// 128x128 tile / WG, K staged 64 at a time, double-buffered via
// GLOBAL_LOAD_ASYNC_TO_LDS_B128 (ASYNCcnt) when available.
// ---------------------------------------------------------------------------
#define MT  128
#define NT  128
#define KT  64
#define LDA 72      // bf16 elems per LDS row (144 B, 16B aligned, bank-padded)

struct SharedU {
    union {
        struct { __bf16 A[2][MT * LDA]; __bf16 B[2][NT * LDA]; } g;  // 73728 B
        float hidden[MT * NT];                                       // 65536 B
    } u;
};

__global__ void __launch_bounds__(256) ffn_gemm_kernel(
    const bf16_t* __restrict__ xb, const bf16_t* __restrict__ wib,
    const float* __restrict__ bi, const float* __restrict__ t_up,
    const float* __restrict__ B_up, const float* __restrict__ A_down,
    float* __restrict__ s_part) {
    __shared__ SharedU sm;

    const int tid    = threadIdx.x;
    const int lane   = tid & 31;
    const int wid    = tid >> 5;
    const int wave_m = wid & 1;    // 0..1 -> 64-row half
    const int wave_n = wid >> 1;   // 0..3 -> 32-col quarter
    const int m_base = blockIdx.y * MT;   // token tile
    const int h_base = blockIdx.x * NT;   // hidden-dim tile

    v8f acc[4][2];
#pragma unroll
    for (int mf = 0; mf < 4; ++mf)
#pragma unroll
        for (int nf = 0; nf < 2; ++nf) acc[mf][nf] = (v8f)(0.0f);

    // fragment-load + 16x WMMA for one staged K-block
    auto compute_stage = [&](int buf) {
        const __bf16* As = &sm.u.g.A[buf][0];
        const __bf16* Bs = &sm.u.g.B[buf][0];
#pragma unroll
        for (int kk = 0; kk < KT; kk += 32) {
            const int klo = kk + ((lane >> 4) << 3);  // lanes 16-31 hold K+8
            v16bf a[4], b[2];
#pragma unroll
            for (int mf = 0; mf < 4; ++mf) {
                const int row = wave_m * 64 + mf * 16 + (lane & 15);
                B32x8 t;
                t.lo = *(const uint4*)&As[row * LDA + klo];
                t.hi = *(const uint4*)&As[row * LDA + klo + 16];
                a[mf] = __builtin_bit_cast(v16bf, t);
            }
#pragma unroll
            for (int nf = 0; nf < 2; ++nf) {
                const int row = wave_n * 32 + nf * 16 + (lane & 15);
                B32x8 t;
                t.lo = *(const uint4*)&Bs[row * LDA + klo];
                t.hi = *(const uint4*)&Bs[row * LDA + klo + 16];
                b[nf] = __builtin_bit_cast(v16bf, t);
            }
#pragma unroll
            for (int mf = 0; mf < 4; ++mf)
#pragma unroll
                for (int nf = 0; nf < 2; ++nf)
                    acc[mf][nf] = __builtin_amdgcn_wmma_f32_16x16x32_bf16(
                        false, a[mf], false, b[nf], (short)0, acc[mf][nf],
                        false, false);
        }
    };

#if USE_ASYNC_LDS
    // async double-buffered staging: overlap next K-block fetch with WMMAs
    auto issue_stage = [&](int k0, int buf) {
#pragma unroll
        for (int i = 0; i < 4; ++i) {
            const int c   = tid + i * 256;
            const int row = c >> 3;
            const int col = (c & 7) * 8;
            __builtin_amdgcn_global_load_async_to_lds_b128(
                (gv4i_p)(xb + (size_t)(m_base + row) * D_IN + k0 + col),
                (lv4i_p)&sm.u.g.A[buf][row * LDA + col], 0, 0);
            __builtin_amdgcn_global_load_async_to_lds_b128(
                (gv4i_p)(wib + (size_t)(h_base + row) * D_IN + k0 + col),
                (lv4i_p)&sm.u.g.B[buf][row * LDA + col], 0, 0);
        }
    };

    issue_stage(0, 0);
    __builtin_amdgcn_s_wait_asynccnt(0);
    __syncthreads();

    int buf = 0;
    for (int k0 = 0; k0 < D_IN; k0 += KT) {
        if (k0 + KT < D_IN) issue_stage(k0 + KT, buf ^ 1);
        compute_stage(buf);
        __builtin_amdgcn_s_wait_asynccnt(0);
        __syncthreads();
        buf ^= 1;
    }
#else
    // synchronous fallback: single buffer, barrier-staged
    for (int k0 = 0; k0 < D_IN; k0 += KT) {
        __syncthreads();
#pragma unroll
        for (int i = 0; i < 4; ++i) {
            const int c   = tid + i * 256;
            const int row = c >> 3;
            const int col = (c & 7) * 8;
            *(v8bf*)&sm.u.g.A[0][row * LDA + col] =
                *(const v8bf*)(xb + (size_t)(m_base + row) * D_IN + k0 + col);
            *(v8bf*)&sm.u.g.B[0][row * LDA + col] =
                *(const v8bf*)(wib + (size_t)(h_base + row) * D_IN + k0 + col);
        }
        __syncthreads();
        compute_stage(0);
    }
#endif
    __syncthreads();  // LDS A/B dead; safe to alias as hidden tile

    // Epilogue: bias + exact GELU + LoRA-up, write hidden tile (xor-swizzled)
#pragma unroll
    for (int nf = 0; nf < 2; ++nf) {
        const int n_local = wave_n * 32 + nf * 16 + (lane & 15);
        const int h       = h_base + n_local;
        const float bias  = bi[h];
        const float4 bu0  = *(const float4*)(B_up + (size_t)h * 8);
        const float4 bu1  = *(const float4*)(B_up + (size_t)h * 8 + 4);
#pragma unroll
        for (int mf = 0; mf < 4; ++mf) {
#pragma unroll
            for (int v = 0; v < 8; ++v) {
                const int m_local = wave_m * 64 + mf * 16 + ((lane & 16) ? 8 : 0) + v;
                const int token   = m_base + m_local;
                float val = acc[mf][nf][v] + bias;
                val = gelu_exact(val);
                const float4 t0 = *(const float4*)(t_up + (size_t)token * 8);
                const float4 t1 = *(const float4*)(t_up + (size_t)token * 8 + 4);
                val += SCALE_F * (t0.x * bu0.x + t0.y * bu0.y + t0.z * bu0.z + t0.w * bu0.w +
                                  t1.x * bu1.x + t1.y * bu1.y + t1.z * bu1.z + t1.w * bu1.w);
                sm.u.hidden[m_local * NT + (n_local ^ (m_local & 63))] = val;
            }
        }
    }
    __syncthreads();

    // Rank-8 contraction of this 128-wide h block: s_part[ht][token][r]
    if (tid < MT) {
        float p[8] = {0.f, 0.f, 0.f, 0.f, 0.f, 0.f, 0.f, 0.f};
        const int sw = tid & 63;
        for (int hh = 0; hh < NT; ++hh) {
            const float hv = sm.u.hidden[tid * NT + (hh ^ sw)];
#pragma unroll
            for (int r = 0; r < 8; ++r)
                p[r] += hv * A_down[(size_t)r * H_DIM + h_base + hh];
        }
        float* dst = s_part + ((size_t)blockIdx.x * N_TOK + m_base + tid) * 8;
#pragma unroll
        for (int r = 0; r < 8; ++r) dst[r] = p[r];
    }
}

// ---------------------------------------------------------------------------
// Kernel 4: reduce s_part over 32 h-tiles, apply B_down, add base, layernorm.
// ---------------------------------------------------------------------------
__global__ void __launch_bounds__(256) finalize_kernel(
    const float* __restrict__ base, const float* __restrict__ B_down,
    const float* __restrict__ s_part, float* __restrict__ out) {
    const int token = blockIdx.x;
    const int tid   = threadIdx.x;

    __shared__ float  sred[256];
    __shared__ float  sd[8];
    __shared__ float2 red2[256];

    {   // gather 32 h-tile partials x 8 ranks
        const int ht = tid >> 3, r = tid & 7;
        sred[tid] = s_part[((size_t)ht * N_TOK + token) * 8 + r];
    }
    __syncthreads();
    if (tid < 8) {
        float s = 0.f;
        for (int ht = 0; ht < 32; ++ht) s += sred[ht * 8 + tid];
        sd[tid] = s * SCALE_F;
    }
    __syncthreads();

    float sdl[8];
#pragma unroll
    for (int r = 0; r < 8; ++r) sdl[r] = sd[r];

    const int d0 = tid * 4;
    const float4 bv = *(const float4*)(base + (size_t)token * D_IN + d0);
    const float bvv[4] = {bv.x, bv.y, bv.z, bv.w};
    float y[4];
    float sum = 0.f, sumsq = 0.f;
#pragma unroll
    for (int j = 0; j < 4; ++j) {
        const float* br = B_down + (size_t)(d0 + j) * 8;
        const float4 b0 = *(const float4*)br;
        const float4 b1 = *(const float4*)(br + 4);
        const float lv = b0.x * sdl[0] + b0.y * sdl[1] + b0.z * sdl[2] + b0.w * sdl[3] +
                         b1.x * sdl[4] + b1.y * sdl[5] + b1.z * sdl[6] + b1.w * sdl[7];
        const float yy = bvv[j] + lv;
        y[j] = yy; sum += yy; sumsq += yy * yy;
    }

    red2[tid] = make_float2(sum, sumsq);
    __syncthreads();
#pragma unroll
    for (int s = 128; s > 0; s >>= 1) {
        if (tid < s) {
            red2[tid].x += red2[tid + s].x;
            red2[tid].y += red2[tid + s].y;
        }
        __syncthreads();
    }
    const float mean = red2[0].x * (1.0f / (float)D_IN);
    const float var  = red2[0].y * (1.0f / (float)D_IN) - mean * mean;
    const float inv  = rsqrtf(var + LN_EPS);

    float4 o;
    o.x = (y[0] - mean) * inv;
    o.y = (y[1] - mean) * inv;
    o.z = (y[2] - mean) * inv;
    o.w = (y[3] - mean) * inv;
    *(float4*)(out + (size_t)token * D_IN + d0) = o;
}

// ---------------------------------------------------------------------------
extern "C" void kernel_launch(void* const* d_in, const int* in_sizes, int n_in,
                              void* d_out, int out_size, void* d_ws, size_t ws_size,
                              hipStream_t stream) {
    const float* x    = (const float*)d_in[0];
    const float* base = (const float*)d_in[1];
    const float* Wi   = (const float*)d_in[2];
    const float* bi   = (const float*)d_in[3];
    const float* A_up = (const float*)d_in[4];
    const float* B_up = (const float*)d_in[5];
    const float* A_dn = (const float*)d_in[6];
    const float* B_dn = (const float*)d_in[7];
    float* out = (float*)d_out;

    char* ws = (char*)d_ws;
    bf16_t* xb     = (bf16_t*)(ws);                       // 32 MB
    bf16_t* wib    = (bf16_t*)(ws + (size_t)33554432);    //  8 MB
    float*  t_up   = (float*)(ws + (size_t)41943040);     // 512 KB
    float*  s_part = (float*)(ws + (size_t)42467328);     // 16 MB

    prep_x_kernel<<<N_TOK, 256, 0, stream>>>(x, A_up, xb, t_up);
    conv_bf16_kernel<<<(H_DIM * D_IN) / (256 * 8), 256, 0, stream>>>(Wi, wib);

    dim3 grid(H_DIM / NT, N_TOK / MT);   // 32 x 128 = 4096 workgroups
    ffn_gemm_kernel<<<grid, 256, 0, stream>>>(xb, wib, bi, t_up, B_up, A_dn, s_part);

    finalize_kernel<<<N_TOK, 256, 0, stream>>>(base, B_dn, s_part, out);
}
